// MultiHeadAttention_9509057593416
// MI455X (gfx1250) — matrix operand order repaired and checked by reading, not compiled, not run on hardware
//
#include <hip/hip_runtime.h>

typedef __attribute__((ext_vector_type(16))) _Float16 v16h;
typedef __attribute__((ext_vector_type(8)))  _Float16 v8h;
typedef __attribute__((ext_vector_type(8)))  float    v8f;
typedef __attribute__((ext_vector_type(4)))  int      v4i;

#define BATCH 2
#define SEQ   2048
#define EMB   1024
#define HEADS 16
#define DHEAD 64
#define ROWS  (BATCH * SEQ)   /* 4096 */
#define C3    (3 * EMB)       /* 3072 */

// ---- gfx1250 async global->LDS copy (guarded; falls back to direct loads) ----
#if defined(__has_builtin)
#if __has_builtin(__builtin_amdgcn_global_load_async_to_lds_b128) && \
    __has_builtin(__builtin_amdgcn_s_wait_asynccnt)
#define USE_ASYNC_LDS 1
#endif
#endif
#ifndef USE_ASYNC_LDS
#define USE_ASYNC_LDS 0
#endif

#if USE_ASYNC_LDS
typedef __attribute__((address_space(1))) v4i* gv4i_p;  // global int4*
typedef __attribute__((address_space(3))) v4i* lv4i_p;  // LDS int4*
__device__ __forceinline__ void async_copy16(const _Float16* g, _Float16* l) {
  // (global src, lds dst, imm offset, cpol); per-lane 16B, tracked by ASYNCcnt
  __builtin_amdgcn_global_load_async_to_lds_b128(
      (gv4i_p)(v4i*)g, (lv4i_p)(v4i*)l, 0, 0);
}
#endif

__device__ __forceinline__ v8f wmma_f16(v16h a, v16h b, v8f c) {
  // D = A(16x32 f16) * B(32x16 f16) + C(16x16 f32)
  return __builtin_amdgcn_wmma_f32_16x16x32_f16(
      false, a, false, b, (short)0, c, false, false);
}

// A fragment (16-bit, 16x32) as built here: lane<16 holds K 0..15, lane>=16 holds K 16..31 (contiguous).
__device__ __forceinline__ v16h make_a_frag(v8h lo, v8h hi) {
  return __builtin_shufflevector(lo, hi, 0,1,2,3,4,5,6,7,8,9,10,11,12,13,14,15);
}

// ---------------------------------------------------------------------------
// fp32 -> f16 elementwise convert
// ---------------------------------------------------------------------------
__global__ void cvt_plain(const float* __restrict__ in,
                          _Float16* __restrict__ out, int total) {
  int idx = blockIdx.x * blockDim.x + threadIdx.x;
  if (idx < total) out[idx] = (_Float16)in[idx];
}

// ---------------------------------------------------------------------------
// fp32 -> f16 transpose: in [K][N] row-major -> out [N][K] row-major
// ---------------------------------------------------------------------------
__global__ void cvt_transpose(const float* __restrict__ in,
                              _Float16* __restrict__ out, int K, int N) {
  int idx = blockIdx.x * blockDim.x + threadIdx.x;
  if (idx < K * N) {
    int k = idx / N, n = idx - k * N;
    out[(size_t)n * K + k] = (_Float16)in[idx];
  }
}

// ---------------------------------------------------------------------------
// QKV projection: [4096,1024](f16) x [1024,3072] + bias.
// One wave per 16(M)x64(N) tile; wT pre-transposed [3072][1024] f16.
// B-panel (64x32, 4KB) staged in LDS, double-buffered async copies with
// compile-time-constant stage indices (k-loop unrolled by 2).
// Q,K written [B,H,N,D]; V written transposed [B,H,D,N] for the PV stage.
// ---------------------------------------------------------------------------
__global__ __launch_bounds__(32) void qkv_gemm(
    const _Float16* __restrict__ xh, const _Float16* __restrict__ wT,
    const float* __restrict__ bias,
    _Float16* __restrict__ qout, _Float16* __restrict__ kout,
    _Float16* __restrict__ vTout) {
  const int lane    = threadIdx.x & 31;
  const int colLane = lane & 15;
  const int hi      = lane >> 4;
  const int m0      = blockIdx.y << 4;
  const int n0      = blockIdx.x << 6;   // 64-wide N tile

  const _Float16* arow = xh + (size_t)(m0 + colLane) * EMB + hi * 16;

  v8f acc[4];
#pragma unroll
  for (int j = 0; j < 4; ++j) acc[j] = (v8f){};

#if USE_ASYNC_LDS
  __shared__ __align__(32) _Float16 Bsh[2][64 * 32];  // [stage][col*32 + k]
  const int crow = lane >> 2;            // 0..7
  const int coff = (lane & 3) * 8;       // halves within a 32-half row
  auto issueB = [&](int kk, int stage) {
#pragma unroll
    for (int c = 0; c < 8; ++c) {
      const int col = c * 8 + crow;
      async_copy16(wT + (size_t)(n0 + col) * EMB + kk + coff,
                   &Bsh[stage][col * 32 + coff]);
    }
  };
  auto compute = [&](int kk, int stage) {
    v16h a = make_a_frag(*(const v8h*)(arow + kk), *(const v8h*)(arow + kk + 8));
#pragma unroll
    for (int j = 0; j < 4; ++j) {
      v16h b = *(const v16h*)&Bsh[stage][(j * 16 + colLane) * 32 + hi * 16];
      acc[j] = wmma_f16(a, b, acc[j]);
    }
  };
  issueB(0, 0);
  for (int kk = 0; kk < EMB; kk += 64) {
    // body A: compute stage 0, prefetch stage 1 (kk+32 always < EMB here)
    issueB(kk + 32, 1);
    __builtin_amdgcn_s_wait_asynccnt(8);
    compute(kk, 0);
    // body B: compute stage 1, prefetch stage 0
    if (kk + 64 < EMB) {
      issueB(kk + 64, 0);
      __builtin_amdgcn_s_wait_asynccnt(8);
    } else {
      __builtin_amdgcn_s_wait_asynccnt(0);
    }
    compute(kk + 32, 1);
  }
#else
  for (int kk = 0; kk < EMB; kk += 32) {
    v16h a = make_a_frag(*(const v8h*)(arow + kk), *(const v8h*)(arow + kk + 8));
#pragma unroll
    for (int j = 0; j < 4; ++j) {
      const _Float16* brow =
          wT + (size_t)(n0 + j * 16 + colLane) * EMB + hi * 16 + kk;
      acc[j] = wmma_f16(a, *(const v16h*)brow, acc[j]);
    }
  }
#endif

#pragma unroll
  for (int j = 0; j < 4; ++j) {
    const int   cg    = n0 + j * 16 + colLane;   // [0,3072)
    const float bs    = bias[cg];
    const int   which = cg >> 10;                // 0=q 1=k 2=v
    const int   c2    = cg & (EMB - 1);
    const int   h     = c2 >> 6;
    const int   d     = c2 & (DHEAD - 1);
#pragma unroll
    for (int r = 0; r < 8; ++r) {
      const int row = m0 + r + hi * 8;
      const int bb  = row >> 11;
      const int n   = row & (SEQ - 1);
      const _Float16 val = (_Float16)(acc[j][r] + bs);
      if (which == 0)
        qout[(((size_t)(bb * HEADS + h)) * SEQ + n) * DHEAD + d] = val;
      else if (which == 1)
        kout[(((size_t)(bb * HEADS + h)) * SEQ + n) * DHEAD + d] = val;
      else
        vTout[(((size_t)(bb * HEADS + h)) * DHEAD + d) * SEQ + n] = val;
    }
  }
}

// ---------------------------------------------------------------------------
// Flash attention: one wave = one 16-query tile of one (b,h); stream keys in
// 32-wide tiles with online softmax.  Q/K are [B,H,N,D]; V is [B,H,D,N] so
// PV B-fragments are single contiguous v16h loads.  Output [B*N, C] f16.
// Softmax uses the whole wave: lanes (l, l+16) split row l's 32 columns and
// combine partial max/sum with a cross-half shuffle.
// ---------------------------------------------------------------------------
__global__ __launch_bounds__(32) void flash_attn(
    const _Float16* __restrict__ q, const _Float16* __restrict__ k,
    const _Float16* __restrict__ vT, _Float16* __restrict__ o) {
  __shared__ __align__(32) float    Sbuf[16][32];
  __shared__ __align__(32) _Float16 Pbuf[16][32];
  __shared__ float alphaBuf[16];
  __shared__ float lBuf[16];

  const int lane    = threadIdx.x & 31;
  const int colLane = lane & 15;
  const int hi      = lane >> 4;
  const int bh      = blockIdx.y;
  const int n0      = blockIdx.x << 4;

  const _Float16* Qb = q  + (size_t)bh * SEQ * DHEAD;
  const _Float16* Kb = k  + (size_t)bh * SEQ * DHEAD;
  const _Float16* Vb = vT + (size_t)bh * DHEAD * SEQ;

  v16h qa[2];
#pragma unroll
  for (int t = 0; t < 2; ++t) {
    const _Float16* qp = Qb + (size_t)(n0 + colLane) * DHEAD + t * 32 + hi * 16;
    qa[t] = make_a_frag(*(const v8h*)qp, *(const v8h*)(qp + 8));
  }

  const float scale = 0.125f;  // 1/sqrt(64)
  v8f oacc[4];
#pragma unroll
  for (int j = 0; j < 4; ++j) oacc[j] = (v8f){};
  // Row state for row = colLane, replicated on lanes l and l+16.
  float mrow = -3.0e38f, lrow = 0.0f;

  for (int kk0 = 0; kk0 < SEQ; kk0 += 32) {
    // ---- S = scale * Q K^T (16x32 tile = two 16x16 WMMA outputs) ----
#pragma unroll
    for (int g = 0; g < 2; ++g) {
      v8f s = {};
#pragma unroll
      for (int t = 0; t < 2; ++t) {
        const _Float16* kp =
            Kb + (size_t)(kk0 + g * 16 + colLane) * DHEAD + t * 32 + hi * 16;
        s = wmma_f16(qa[t], *(const v16h*)kp, s);
      }
#pragma unroll
      for (int r = 0; r < 8; ++r)
        Sbuf[r + hi * 8][g * 16 + colLane] = s[r] * scale;
    }
    __syncthreads();

    // ---- online softmax: row = colLane, this lane covers cols hi*16..+16 ----
    {
      float pmax = -3.0e38f;
#pragma unroll
      for (int c = 0; c < 16; ++c)
        pmax = fmaxf(pmax, Sbuf[colLane][hi * 16 + c]);
      const float tmax = fmaxf(fmaxf(pmax, __shfl_xor(pmax, 16, 32)), mrow);
      const float alpha = __expf(mrow - tmax);
      float psum = 0.0f;
#pragma unroll
      for (int c = 0; c < 16; ++c) {
        float p = __expf(Sbuf[colLane][hi * 16 + c] - tmax);
        Pbuf[colLane][hi * 16 + c] = (_Float16)p;
        psum += p;
      }
      const float tsum = psum + __shfl_xor(psum, 16, 32);
      lrow = lrow * alpha + tsum;
      mrow = tmax;
      if (lane < 16) alphaBuf[lane] = alpha;
    }
    __syncthreads();

    // ---- rescale running O accumulators ----
#pragma unroll
    for (int r = 0; r < 8; ++r) {
      const float a_r = alphaBuf[r + hi * 8];
#pragma unroll
      for (int j = 0; j < 4; ++j) oacc[j][r] *= a_r;
    }

    // ---- O += P V ----
    const _Float16* pp = &Pbuf[colLane][hi * 16];
    v16h pa = make_a_frag(*(const v8h*)pp, *(const v8h*)(pp + 8));
#pragma unroll
    for (int j = 0; j < 4; ++j) {
      const _Float16* vp =
          Vb + (size_t)(j * 16 + colLane) * SEQ + kk0 + hi * 16;
      oacc[j] = wmma_f16(pa, *(const v16h*)vp, oacc[j]);
    }
    __syncthreads();  // protect Sbuf/Pbuf reuse
  }

  if (lane < 16) lBuf[lane] = lrow;
  __syncthreads();

  const int b = bh >> 4;
  const int h = bh & (HEADS - 1);
#pragma unroll
  for (int r = 0; r < 8; ++r) {
    const int    row  = r + hi * 8;
    const float  inv  = 1.0f / lBuf[row];
    const size_t base = ((size_t)(b * SEQ + n0 + row)) * EMB + h * DHEAD;
#pragma unroll
    for (int j = 0; j < 4; ++j)
      o[base + j * 16 + colLane] = (_Float16)(oacc[j][r] * inv);
  }
}

// ---------------------------------------------------------------------------
// Output projection: [4096,1024](f16) x [1024,1024] + bias -> fp32 d_out.
// One wave per 16x64 tile; B staged via async LDS double buffer (unrolled x2).
// ---------------------------------------------------------------------------
__global__ __launch_bounds__(32) void proj_gemm(
    const _Float16* __restrict__ A, const _Float16* __restrict__ wT,
    const float* __restrict__ bias, float* __restrict__ out) {
  const int lane    = threadIdx.x & 31;
  const int colLane = lane & 15;
  const int hi      = lane >> 4;
  const int m0      = blockIdx.y << 4;
  const int n0      = blockIdx.x << 6;

  const _Float16* arow = A + (size_t)(m0 + colLane) * EMB + hi * 16;

  v8f acc[4];
#pragma unroll
  for (int j = 0; j < 4; ++j) acc[j] = (v8f){};

#if USE_ASYNC_LDS
  __shared__ __align__(32) _Float16 Bsh[2][64 * 32];
  const int crow = lane >> 2;
  const int coff = (lane & 3) * 8;
  auto issueB = [&](int kk, int stage) {
#pragma unroll
    for (int c = 0; c < 8; ++c) {
      const int col = c * 8 + crow;
      async_copy16(wT + (size_t)(n0 + col) * EMB + kk + coff,
                   &Bsh[stage][col * 32 + coff]);
    }
  };
  auto compute = [&](int kk, int stage) {
    v16h a = make_a_frag(*(const v8h*)(arow + kk), *(const v8h*)(arow + kk + 8));
#pragma unroll
    for (int j = 0; j < 4; ++j) {
      v16h b = *(const v16h*)&Bsh[stage][(j * 16 + colLane) * 32 + hi * 16];
      acc[j] = wmma_f16(a, b, acc[j]);
    }
  };
  issueB(0, 0);
  for (int kk = 0; kk < EMB; kk += 64) {
    issueB(kk + 32, 1);
    __builtin_amdgcn_s_wait_asynccnt(8);
    compute(kk, 0);
    if (kk + 64 < EMB) {
      issueB(kk + 64, 0);
      __builtin_amdgcn_s_wait_asynccnt(8);
    } else {
      __builtin_amdgcn_s_wait_asynccnt(0);
    }
    compute(kk + 32, 1);
  }
#else
  for (int kk = 0; kk < EMB; kk += 32) {
    v16h a = make_a_frag(*(const v8h*)(arow + kk), *(const v8h*)(arow + kk + 8));
#pragma unroll
    for (int j = 0; j < 4; ++j) {
      const _Float16* brow =
          wT + (size_t)(n0 + j * 16 + colLane) * EMB + hi * 16 + kk;
      acc[j] = wmma_f16(a, *(const v16h*)brow, acc[j]);
    }
  }
#endif

#pragma unroll
  for (int j = 0; j < 4; ++j) {
    const float bs = bias[n0 + j * 16 + colLane];
#pragma unroll
    for (int r = 0; r < 8; ++r) {
      const int row = m0 + r + hi * 8;
      out[(size_t)row * EMB + n0 + j * 16 + colLane] = acc[j][r] + bs;
    }
  }
}

// ---------------------------------------------------------------------------
extern "C" void kernel_launch(void* const* d_in, const int* in_sizes, int n_in,
                              void* d_out, int out_size, void* d_ws, size_t ws_size,
                              hipStream_t stream) {
  const float* x      = (const float*)d_in[0];
  const float* w_qkv  = (const float*)d_in[1];
  const float* b_qkv  = (const float*)d_in[2];
  const float* w_proj = (const float*)d_in[3];
  const float* b_proj = (const float*)d_in[4];
  float* out = (float*)d_out;

  // Workspace carve-up (f16), ~48 MB total
  _Float16* wqkvT  = (_Float16*)d_ws;                       // [3072][1024]
  _Float16* wprojT = wqkvT  + (size_t)C3  * EMB;            // [1024][1024]
  _Float16* xh     = wprojT + (size_t)EMB * EMB;            // [4096][1024]
  _Float16* qh     = xh + (size_t)ROWS * EMB;               // [B,H,N,D]
  _Float16* kh     = qh + (size_t)BATCH * HEADS * SEQ * DHEAD;
  _Float16* vTh    = kh + (size_t)BATCH * HEADS * SEQ * DHEAD;  // [B,H,D,N]
  _Float16* attn   = vTh + (size_t)BATCH * HEADS * SEQ * DHEAD; // [4096][1024]

  {
    int total = EMB * C3;
    cvt_transpose<<<(total + 255) / 256, 256, 0, stream>>>(w_qkv, wqkvT, EMB, C3);
  }
  {
    int total = EMB * EMB;
    cvt_transpose<<<(total + 255) / 256, 256, 0, stream>>>(w_proj, wprojT, EMB, EMB);
  }
  {
    int total = ROWS * EMB;
    cvt_plain<<<(total + 255) / 256, 256, 0, stream>>>(x, xh, total);
  }
  qkv_gemm<<<dim3(C3 / 64, ROWS / 16), 32, 0, stream>>>(xh, wqkvT, b_qkv, qh, kh, vTh);
  flash_attn<<<dim3(SEQ / 16, BATCH * HEADS), 32, 0, stream>>>(qh, kh, vTh, attn);
  proj_gemm<<<dim3(EMB / 64, ROWS / 16), 32, 0, stream>>>(attn, wprojT, b_proj, out);
}
